// NeuralFieldDiffusion_5927054868648
// MI455X (gfx1250) — compile-verified, matrix operand order checked by reading, not run on hardware
//
#include <hip/hip_runtime.h>
#include <hip/hip_bf16.h>

// ---------------------------------------------------------------------------
// Types for CDNA5 WMMA (wave32): v_wmma_f32_16x16x32_f16
// ---------------------------------------------------------------------------
typedef _Float16 h8v  __attribute__((ext_vector_type(8)));
typedef _Float16 h16v __attribute__((ext_vector_type(16)));
typedef float    f8v  __attribute__((ext_vector_type(8)));

__device__ __forceinline__ h16v load_frag_h16(const _Float16* p) {
    // two 16-byte loads: elements 0..7 = K-low half, 8..15 = K-high half
    union { h16v v; h8v h[2]; } u;
    u.h[0] = *(const h8v*)(p);
    u.h[1] = *(const h8v*)(p + 16);
    return u.v;
}

__device__ __forceinline__ h16v load_frag_f32cvt(const float* p) {
    union { h16v v; h8v h[2]; } u;
    float4 a0 = *(const float4*)(p);
    float4 a1 = *(const float4*)(p + 4);
    float4 a2 = *(const float4*)(p + 16);
    float4 a3 = *(const float4*)(p + 20);
    u.v[0]=(_Float16)a0.x;  u.v[1]=(_Float16)a0.y;  u.v[2]=(_Float16)a0.z;  u.v[3]=(_Float16)a0.w;
    u.v[4]=(_Float16)a1.x;  u.v[5]=(_Float16)a1.y;  u.v[6]=(_Float16)a1.z;  u.v[7]=(_Float16)a1.w;
    u.v[8]=(_Float16)a2.x;  u.v[9]=(_Float16)a2.y;  u.v[10]=(_Float16)a2.z; u.v[11]=(_Float16)a2.w;
    u.v[12]=(_Float16)a3.x; u.v[13]=(_Float16)a3.y; u.v[14]=(_Float16)a3.z; u.v[15]=(_Float16)a3.w;
    return u.v;
}

__device__ __forceinline__ float gelu_f(float x) {
    return 0.5f * x * (1.0f + erff(x * 0.70710678118654752f));  // exact GELU
}

// ---------------------------------------------------------------------------
// Generic wave-level GEMM on LDS-staged activations.
//   act: row-major f16 LDS, row stride = STRIDE halfs (16B-aligned rows)
//   Each wave owns rows [rowbase, rowbase+16): computes all NOUT columns,
//   applies bias (+ optional LayerNorm, + optional GELU) in fragment layout,
//   writes f16 result back in-place (wave-private rows -> no barrier needed).
//   W: global f16 weights, row-major [NOUT][K]  (B[k][n] = W[n][k]).
// ---------------------------------------------------------------------------
template<int K, int NOUT, int STRIDE, bool DO_LN, bool DO_GELU>
__device__ __forceinline__ void wave_gemm(_Float16* act, int rowbase,
                                          const _Float16* __restrict__ W,
                                          const float* __restrict__ bias,
                                          const float* __restrict__ gamma,
                                          const float* __restrict__ beta,
                                          int lane)
{
    constexpr int KS = K / 32;
    constexpr int NT = NOUT / 16;
    const int m     = lane & 15;           // row (A) / col (B,D) within tile
    const int off8  = (lane >> 4) << 3;    // K-offset select for high lanes
    const int half8 = (lane >> 4) << 3;    // D-layout row offset for high lanes

    // --- A fragments: 16 rows x K, hoisted across all N tiles -------------
    h16v af[KS];
    {
        const _Float16* ap = act + (rowbase + m) * STRIDE + off8;
#pragma unroll
        for (int ks = 0; ks < KS; ++ks) af[ks] = load_frag_h16(ap + ks * 32);
    }

    // --- accumulate all N tiles -------------------------------------------
    f8v acc[NT];
#pragma unroll
    for (int t = 0; t < NT; ++t) {
        f8v a; 
#pragma unroll
        for (int r = 0; r < 8; ++r) a[r] = 0.0f;
        const _Float16* bp = W + (size_t)(16 * t + m) * K + off8;
#pragma unroll
        for (int ks = 0; ks < KS; ++ks) {
            h16v bf = load_frag_h16(bp + ks * 32);
            a = __builtin_amdgcn_wmma_f32_16x16x32_f16(false, af[ks], false, bf,
                                                       (short)0, a, false, false);
        }
        acc[t] = a;
    }

    // --- bias --------------------------------------------------------------
#pragma unroll
    for (int t = 0; t < NT; ++t) {
        float bv = bias[16 * t + m];
#pragma unroll
        for (int r = 0; r < 8; ++r) acc[t][r] += bv;
    }

    // --- LayerNorm in fragment layout (row stats via 16-lane xor-reduce) ---
    if (DO_LN) {
        float mean[8], rstd[8];
#pragma unroll
        for (int r = 0; r < 8; ++r) {
            float s = 0.f, q = 0.f;
#pragma unroll
            for (int t = 0; t < NT; ++t) { float x = acc[t][r]; s += x; q += x * x; }
            for (int off = 1; off <= 8; off <<= 1) {
                s += __shfl_xor(s, off, 32);
                q += __shfl_xor(q, off, 32);
            }
            float mu  = s * (1.0f / (float)NOUT);
            float var = q * (1.0f / (float)NOUT) - mu * mu;
            mean[r] = mu;
            rstd[r] = rsqrtf(var + 1e-5f);
        }
#pragma unroll
        for (int t = 0; t < NT; ++t) {
            float g  = gamma[16 * t + m];
            float be = beta[16 * t + m];
#pragma unroll
            for (int r = 0; r < 8; ++r)
                acc[t][r] = (acc[t][r] - mean[r]) * rstd[r] * g + be;
        }
    }

    if (DO_GELU) {
#pragma unroll
        for (int t = 0; t < NT; ++t)
#pragma unroll
            for (int r = 0; r < 8; ++r) acc[t][r] = gelu_f(acc[t][r]);
    }

    // --- write back (D layout -> row-major f16 LDS, wave-private rows) -----
#pragma unroll
    for (int t = 0; t < NT; ++t)
#pragma unroll
        for (int r = 0; r < 8; ++r)
            act[(rowbase + r + half8) * STRIDE + 16 * t + m] = (_Float16)acc[t][r];
}

// ---------------------------------------------------------------------------
// Kernel 0: weight prep — f32 -> f16 row-major (pad K 3->32), init ctx keys.
// ---------------------------------------------------------------------------
__global__ void prep_kernel(const float* __restrict__ pw1, const float* __restrict__ pw2,
                            const float* __restrict__ pw3, const float* __restrict__ pw4,
                            _Float16* W1p, _Float16* W2h, _Float16* W3h, _Float16* W4h,
                            unsigned* ctx_keys)
{
    int i = blockIdx.x * blockDim.x + threadIdx.x;
    int st = gridDim.x * blockDim.x;
    for (int k = i; k < 128 * 32; k += st) {
        int r = k >> 5, c = k & 31;
        W1p[k] = (_Float16)(c < 3 ? pw1[r * 3 + c] : 0.0f);
    }
    for (int k = i; k < 128 * 128; k += st) W2h[k] = (_Float16)pw2[k];
    for (int k = i; k < 256 * 128; k += st) W3h[k] = (_Float16)pw3[k];
    for (int k = i; k < 256 * 256; k += st) W4h[k] = (_Float16)pw4[k];
    for (int k = i; k < 32 * 256; k += st) ctx_keys[k] = 0u;   // below every real key
}

// ---------------------------------------------------------------------------
// Kernel 1: PointNet encoder. 64 points / workgroup, 4 waves x 16 rows.
// WMMA layers 3->128->128->256->256 with LN(+GELU), per-channel max via
// monotonic-key u32 atomicMax.
// ---------------------------------------------------------------------------
#define PN_STRIDE 264   // 256 cols + 8 pad halfs; 528B row = 33*16B (aligned)

__global__ void __launch_bounds__(128)
pointnet_kernel(const float* __restrict__ points,
                const _Float16* __restrict__ W1p, const float* __restrict__ b1,
                const float* __restrict__ g1, const float* __restrict__ be1,
                const _Float16* __restrict__ W2h, const float* __restrict__ b2,
                const float* __restrict__ g2, const float* __restrict__ be2,
                const _Float16* __restrict__ W3h, const float* __restrict__ b3,
                const float* __restrict__ g3, const float* __restrict__ be3,
                const _Float16* __restrict__ W4h, const float* __restrict__ b4,
                const float* __restrict__ g4, const float* __restrict__ be4,
                unsigned* __restrict__ ctx_keys)
{
    __shared__ __align__(16) _Float16 act[64 * PN_STRIDE];
    const int tid  = threadIdx.x;
    const int wave = tid >> 5;
    const int lane = tid & 31;
    const long long p0 = (long long)blockIdx.x * 64;
    const int b = blockIdx.x >> 8;   // 16384/64 = 256 blocks per batch

    // stage input points, K padded 3 -> 32
    if (tid < 64) {
        const float* p = points + (p0 + tid) * 3;
        _Float16* row = act + tid * PN_STRIDE;
        row[0] = (_Float16)p[0]; row[1] = (_Float16)p[1]; row[2] = (_Float16)p[2];
#pragma unroll
        for (int c = 3; c < 32; ++c) row[c] = (_Float16)0.0f;
    }
    __syncthreads();

    const int rowbase = wave * 16;
    wave_gemm< 32, 128, PN_STRIDE, true, true >(act, rowbase, W1p, b1, g1, be1, lane);
    wave_gemm<128, 128, PN_STRIDE, true, true >(act, rowbase, W2h, b2, g2, be2, lane);
    wave_gemm<128, 256, PN_STRIDE, true, true >(act, rowbase, W3h, b3, g3, be3, lane);
    wave_gemm<256, 256, PN_STRIDE, true, false>(act, rowbase, W4h, b4, g4, be4, lane);
    __syncthreads();

    // per-channel max over the 64 local rows, merged globally with u32 keys
    for (int col = tid; col < 256; col += 128) {
        float m = -3.4e38f;
        for (int r = 0; r < 64; ++r)
            m = fmaxf(m, (float)act[r * PN_STRIDE + col]);
        unsigned u   = __float_as_uint(m);
        unsigned key = (u & 0x80000000u) ? ~u : (u | 0x80000000u);  // monotonic
        atomicMax(ctx_keys + b * 256 + col, key);
    }
}

// ---------------------------------------------------------------------------
// Kernel 2: decode ctx, time embedding, hypernet projection -> c [32,256].
// One block per batch, 256 threads.
// ---------------------------------------------------------------------------
__device__ __forceinline__ float block_sum_256(float v, float* s_red, int t) {
    s_red[t] = v; __syncthreads();
    for (int off = 128; off > 0; off >>= 1) {
        if (t < off) s_red[t] += s_red[t + off];
        __syncthreads();
    }
    float r = s_red[0]; __syncthreads();
    return r;
}

__global__ void __launch_bounds__(256)
ctx_kernel(const unsigned* __restrict__ keys, const float* __restrict__ tarr,
           const float* __restrict__ hw1, const float* __restrict__ hb1,
           const float* __restrict__ hg1, const float* __restrict__ hbe1,
           const float* __restrict__ hw2, const float* __restrict__ hb2,
           const float* __restrict__ hg2, const float* __restrict__ hbe2,
           float* __restrict__ c32, _Float16* __restrict__ c16)
{
    __shared__ float s_in[320];
    __shared__ float s_mid[256];
    __shared__ float s_red[256];
    const int b = blockIdx.x, t = threadIdx.x;

    {   // decode per-channel max keys
        unsigned key = keys[b * 256 + t];
        unsigned u = (key & 0x80000000u) ? (key & 0x7fffffffu) : ~key;
        s_in[t] = __uint_as_float(u);
    }
    if (t < 32) {   // sinusoidal time embedding, half = 32
        float e = expf((float)t * (-9.210340371976184f / 31.0f)); // -ln(1e4)/31
        float a = tarr[b] * e;
        s_in[256 + t] = sinf(a);
        s_in[288 + t] = cosf(a);
    }
    __syncthreads();

    // hp layer 1: 320 -> 256, LN + GELU
    float s = hb1[t];
    {
        const float* wr = hw1 + t * 320;
        for (int k = 0; k < 320; ++k) s += s_in[k] * wr[k];
    }
    float sum = block_sum_256(s, s_red, t);
    float sq  = block_sum_256(s * s, s_red, t);
    float mu  = sum * (1.0f / 256.0f);
    float var = sq * (1.0f / 256.0f) - mu * mu;
    float h   = gelu_f((s - mu) * rsqrtf(var + 1e-5f) * hg1[t] + hbe1[t]);
    s_mid[t] = h; __syncthreads();

    // hp layer 2: 256 -> 256, LN + GELU
    float s2 = hb2[t];
    {
        const float* wr = hw2 + t * 256;
        for (int k = 0; k < 256; ++k) s2 += s_mid[k] * wr[k];
    }
    sum = block_sum_256(s2, s_red, t);
    sq  = block_sum_256(s2 * s2, s_red, t);
    mu  = sum * (1.0f / 256.0f);
    var = sq * (1.0f / 256.0f) - mu * mu;
    float c = gelu_f((s2 - mu) * rsqrtf(var + 1e-5f) * hg2[t] + hbe2[t]);
    c32[b * 256 + t] = c;
    c16[b * 256 + t] = (_Float16)c;
}

// ---------------------------------------------------------------------------
// Kernel 3: weight-generator GEMM via WMMA:  out[32][Nout] = c @ Wg^T + bg.
// 2 waves/block (batches 0-15 / 16-31), 16 output columns per block.
// Generator weights converted f32 -> f16 on the fly (they stream from L2).
// ---------------------------------------------------------------------------
__global__ void __launch_bounds__(64)
wgen_kernel(const _Float16* __restrict__ c16, const float* __restrict__ wgw,
            const float* __restrict__ wgb, float* __restrict__ out, int Nout)
{
    const int lane = threadIdx.x & 31;
    const int wave = threadIdx.x >> 5;
    const int rowbase = wave * 16;
    const int n0 = blockIdx.x * 16;
    const int m     = lane & 15;
    const int off8  = (lane >> 4) << 3;
    const int half8 = (lane >> 4) << 3;

    h16v af[8];
    {
        const _Float16* ap = c16 + (rowbase + m) * 256 + off8;
#pragma unroll
        for (int ks = 0; ks < 8; ++ks) af[ks] = load_frag_h16(ap + ks * 32);
    }
    const int n = n0 + m;
    const float* bp = wgw + (size_t)n * 256 + off8;
    f8v a;
#pragma unroll
    for (int r = 0; r < 8; ++r) a[r] = 0.0f;
#pragma unroll
    for (int ks = 0; ks < 8; ++ks) {
        h16v bf = load_frag_f32cvt(bp + ks * 32);
        a = __builtin_amdgcn_wmma_f32_16x16x32_f16(false, af[ks], false, bf,
                                                   (short)0, a, false, false);
    }
    float bias = wgb[n];
#pragma unroll
    for (int r = 0; r < 8; ++r)
        out[(size_t)(rowbase + r + half8) * Nout + n] = a[r] + bias;
}

// ---------------------------------------------------------------------------
// Kernel 3b: bias generators (tiny): bws{0,1,2}[b] = c[b] @ bgw^T + bgb.
// ---------------------------------------------------------------------------
__global__ void __launch_bounds__(128)
biasgen_kernel(const float* __restrict__ c32,
               const float* __restrict__ bg0w, const float* __restrict__ bg0b,
               const float* __restrict__ bg1w, const float* __restrict__ bg1b,
               const float* __restrict__ bg2w, const float* __restrict__ bg2b,
               float* __restrict__ bws0, float* __restrict__ bws1, float* __restrict__ bws2)
{
    __shared__ float sc[256];
    const int b = blockIdx.x, t = threadIdx.x;
    sc[t] = c32[b * 256 + t];
    sc[t + 128] = c32[b * 256 + t + 128];
    __syncthreads();

    {
        float s = bg0b[t]; const float* wr = bg0w + t * 256;
        for (int k = 0; k < 256; ++k) s += sc[k] * wr[k];
        bws0[b * 128 + t] = s;
    }
    {
        float s = bg1b[t]; const float* wr = bg1w + t * 256;
        for (int k = 0; k < 256; ++k) s += sc[k] * wr[k];
        bws1[b * 128 + t] = s;
    }
    if (t < 16) {
        float s = 0.0f;
        if (t < 3) {
            s = bg2b[t]; const float* wr = bg2w + t * 256;
            for (int k = 0; k < 256; ++k) s += sc[k] * wr[k];
        }
        bws2[b * 16 + t] = s;
    }
}

// ---------------------------------------------------------------------------
// Kernel 4: F.normalize generated weight rows, emit padded f16 per-batch
// field weights: w0p [B][128][64], w1p [B][128][128], w2p [B][16][128].
// grid (32, 272), one wave per (batch, row-slot).
// ---------------------------------------------------------------------------
__global__ void __launch_bounds__(32)
normalize_kernel(const float* __restrict__ wr0, const float* __restrict__ wr1,
                 const float* __restrict__ wr2,
                 _Float16* __restrict__ w0p, _Float16* __restrict__ w1p,
                 _Float16* __restrict__ w2p)
{
    const int b = blockIdx.x;
    const int slot = blockIdx.y;
    const int lane = threadIdx.x;

    const float* src = nullptr;
    _Float16* dst;
    int din, padK;
    bool zero = false;
    if (slot < 128) {
        din = 63; padK = 64;
        src = wr0 + b * 8064 + slot * 63;
        dst = w0p + (size_t)(b * 128 + slot) * 64;
    } else if (slot < 256) {
        int o = slot - 128; din = 128; padK = 128;
        src = wr1 + b * 16384 + o * 128;
        dst = w1p + (size_t)(b * 128 + o) * 128;
    } else {
        int o = slot - 256; din = 128; padK = 128;
        dst = w2p + (size_t)(b * 16 + o) * 128;
        if (o < 3) src = wr2 + b * 384 + o * 128; else zero = true;
    }

    float q = 0.0f;
    if (!zero)
        for (int k = lane; k < din; k += 32) { float v = src[k]; q += v * v; }
    for (int off = 16; off >= 1; off >>= 1) q += __shfl_xor(q, off, 32);
    float scale = zero ? 0.0f : 1.0f / fmaxf(sqrtf(q), 1e-12f);
    for (int k = lane; k < padK; k += 32)
        dst[k] = (_Float16)((!zero && k < din) ? src[k] * scale : 0.0f);
}

// ---------------------------------------------------------------------------
// Kernel 5: field MLP. 64 points / workgroup; Fourier PE -> 128 -> 128 -> 3.
// ---------------------------------------------------------------------------
#define FD_STRIDE 136   // 128 cols + 8 pad halfs; 272B row = 17*16B (aligned)

__device__ __forceinline__ void wave_field_out(const _Float16* act, int rowbase,
                                               const _Float16* __restrict__ W,
                                               const float* __restrict__ bias,
                                               float* __restrict__ out,
                                               long long p0, int lane)
{
    const int m     = lane & 15;
    const int off8  = (lane >> 4) << 3;
    const int half8 = (lane >> 4) << 3;
    h16v af[4];
    {
        const _Float16* ap = act + (rowbase + m) * FD_STRIDE + off8;
#pragma unroll
        for (int ks = 0; ks < 4; ++ks) af[ks] = load_frag_h16(ap + ks * 32);
    }
    const _Float16* bp = W + (size_t)m * 128 + off8;
    f8v a;
#pragma unroll
    for (int r = 0; r < 8; ++r) a[r] = 0.0f;
#pragma unroll
    for (int ks = 0; ks < 4; ++ks) {
        h16v bf = load_frag_h16(bp + ks * 32);
        a = __builtin_amdgcn_wmma_f32_16x16x32_f16(false, af[ks], false, bf,
                                                   (short)0, a, false, false);
    }
    if (m < 3) {
        float bv = bias[m];
#pragma unroll
        for (int r = 0; r < 8; ++r)
            out[(p0 + rowbase + r + half8) * 3 + m] = a[r] + bv;
    }
}

__global__ void __launch_bounds__(128)
field_kernel(const float* __restrict__ points,
             const _Float16* __restrict__ w0p, const float* __restrict__ bws0,
             const _Float16* __restrict__ w1p, const float* __restrict__ bws1,
             const _Float16* __restrict__ w2p, const float* __restrict__ bws2,
             float* __restrict__ out)
{
    __shared__ __align__(16) _Float16 act[64 * FD_STRIDE];
    const int tid  = threadIdx.x;
    const int wave = tid >> 5;
    const int lane = tid & 31;
    const long long p0 = (long long)blockIdx.x * 64;
    const int b = blockIdx.x >> 8;

    // Fourier positional encoding: [xyz(3), sin(30), cos(30), pad] -> 64 cols
    {
        const int row  = tid & 63;
        const int part = tid >> 6;           // 0: cols 0..31, 1: cols 32..63
        const float* p = points + (p0 + row) * 3;
        float pv[3] = { p[0], p[1], p[2] };
        _Float16* r = act + row * FD_STRIDE;
        for (int c = part * 32; c < part * 32 + 32; ++c) {
            float v;
            if (c < 3) v = pv[c];
            else if (c < 33) {
                int j = c - 3; int d = j / 10; int f = j % 10;
                v = sinf(pv[d] * (float)(1 << f) * 3.14159265358979323846f);
            } else if (c < 63) {
                int j = c - 33; int d = j / 10; int f = j % 10;
                v = cosf(pv[d] * (float)(1 << f) * 3.14159265358979323846f);
            } else v = 0.0f;
            r[c] = (_Float16)v;
        }
    }
    __syncthreads();

    const int rowbase = wave * 16;
    wave_gemm< 64, 128, FD_STRIDE, false, true>(act, rowbase, w0p + (size_t)b * 128 * 64,
                                                bws0 + b * 128, nullptr, nullptr, lane);
    wave_gemm<128, 128, FD_STRIDE, false, true>(act, rowbase, w1p + (size_t)b * 128 * 128,
                                                bws1 + b * 128, nullptr, nullptr, lane);
    wave_field_out(act, rowbase, w2p + (size_t)b * 16 * 128, bws2 + b * 16, out, p0, lane);
}

// ---------------------------------------------------------------------------
// Host launcher
// ---------------------------------------------------------------------------
extern "C" void kernel_launch(void* const* d_in, const int* in_sizes, int n_in,
                              void* d_out, int out_size, void* d_ws, size_t ws_size,
                              hipStream_t stream)
{
    (void)in_sizes; (void)n_in; (void)out_size; (void)ws_size;
    const float* points = (const float*)d_in[0];
    const float* tarr   = (const float*)d_in[1];
    const float* pn_w1  = (const float*)d_in[2];
    const float* pn_b1  = (const float*)d_in[3];
    const float* pn_g1  = (const float*)d_in[4];
    const float* pn_be1 = (const float*)d_in[5];
    const float* pn_w2  = (const float*)d_in[6];
    const float* pn_b2  = (const float*)d_in[7];
    const float* pn_g2  = (const float*)d_in[8];
    const float* pn_be2 = (const float*)d_in[9];
    const float* pn_w3  = (const float*)d_in[10];
    const float* pn_b3  = (const float*)d_in[11];
    const float* pn_g3  = (const float*)d_in[12];
    const float* pn_be3 = (const float*)d_in[13];
    const float* pn_w4  = (const float*)d_in[14];
    const float* pn_b4  = (const float*)d_in[15];
    const float* pn_g4  = (const float*)d_in[16];
    const float* pn_be4 = (const float*)d_in[17];
    const float* hp_w1  = (const float*)d_in[18];
    const float* hp_b1  = (const float*)d_in[19];
    const float* hp_g1  = (const float*)d_in[20];
    const float* hp_be1 = (const float*)d_in[21];
    const float* hp_w2  = (const float*)d_in[22];
    const float* hp_b2  = (const float*)d_in[23];
    const float* hp_g2  = (const float*)d_in[24];
    const float* hp_be2 = (const float*)d_in[25];
    const float* wg0_w  = (const float*)d_in[26];
    const float* wg0_b  = (const float*)d_in[27];
    const float* bg0_w  = (const float*)d_in[28];
    const float* bg0_b  = (const float*)d_in[29];
    const float* wg1_w  = (const float*)d_in[30];
    const float* wg1_b  = (const float*)d_in[31];
    const float* bg1_w  = (const float*)d_in[32];
    const float* bg1_b  = (const float*)d_in[33];
    const float* wg2_w  = (const float*)d_in[34];
    const float* wg2_b  = (const float*)d_in[35];
    const float* bg2_w  = (const float*)d_in[36];
    const float* bg2_b  = (const float*)d_in[37];
    float* out = (float*)d_out;

    // ---- workspace carve-out (256B aligned slots) ----
    char* ws = (char*)d_ws;
    size_t cur = 0;
    auto take = [&](size_t bytes) -> void* {
        void* p = ws + cur;
        cur += (bytes + 255) & ~(size_t)255;
        return p;
    };
    _Float16* W1p   = (_Float16*)take(128 * 32  * sizeof(_Float16));
    _Float16* W2h   = (_Float16*)take(128 * 128 * sizeof(_Float16));
    _Float16* W3h   = (_Float16*)take(256 * 128 * sizeof(_Float16));
    _Float16* W4h   = (_Float16*)take(256 * 256 * sizeof(_Float16));
    unsigned* keys  = (unsigned*)take(32 * 256 * sizeof(unsigned));
    float*    c32   = (float*)   take(32 * 256 * sizeof(float));
    _Float16* c16   = (_Float16*)take(32 * 256 * sizeof(_Float16));
    float*    wraw0 = (float*)   take((size_t)32 * 8064  * sizeof(float));
    float*    wraw1 = (float*)   take((size_t)32 * 16384 * sizeof(float));
    float*    wraw2 = (float*)   take((size_t)32 * 384   * sizeof(float));
    _Float16* w0p   = (_Float16*)take((size_t)32 * 128 * 64  * sizeof(_Float16));
    _Float16* w1p   = (_Float16*)take((size_t)32 * 128 * 128 * sizeof(_Float16));
    _Float16* w2p   = (_Float16*)take((size_t)32 * 16  * 128 * sizeof(_Float16));
    float*    bws0  = (float*)   take(32 * 128 * sizeof(float));
    float*    bws1  = (float*)   take(32 * 128 * sizeof(float));
    float*    bws2  = (float*)   take(32 * 16  * sizeof(float));

    // 0: prep (weight f16 conversion + ctx-key init)
    prep_kernel<<<128, 256, 0, stream>>>(pn_w1, pn_w2, pn_w3, pn_w4,
                                         W1p, W2h, W3h, W4h, keys);
    // 1: PointNet encoder (WMMA) + global channel max
    pointnet_kernel<<<8192, 128, 0, stream>>>(points,
        W1p, pn_b1, pn_g1, pn_be1,
        W2h, pn_b2, pn_g2, pn_be2,
        W3h, pn_b3, pn_g3, pn_be3,
        W4h, pn_b4, pn_g4, pn_be4, keys);
    // 2: context decode + time embedding + hypernet projection
    ctx_kernel<<<32, 256, 0, stream>>>(keys, tarr,
        hp_w1, hp_b1, hp_g1, hp_be1,
        hp_w2, hp_b2, hp_g2, hp_be2, c32, c16);
    // 3: generated-weight GEMMs (WMMA)
    wgen_kernel<<<504,  64, 0, stream>>>(c16, wg0_w, wg0_b, wraw0, 8064);
    wgen_kernel<<<1024, 64, 0, stream>>>(c16, wg1_w, wg1_b, wraw1, 16384);
    wgen_kernel<<<24,   64, 0, stream>>>(c16, wg2_w, wg2_b, wraw2, 384);
    // 3b: generated biases
    biasgen_kernel<<<32, 128, 0, stream>>>(c32, bg0_w, bg0_b, bg1_w, bg1_b,
                                           bg2_w, bg2_b, bws0, bws1, bws2);
    // 4: F.normalize rows -> padded per-batch f16 field weights
    normalize_kernel<<<dim3(32, 272), 32, 0, stream>>>(wraw0, wraw1, wraw2,
                                                       w0p, w1p, w2p);
    // 5: field MLP (WMMA) -> velocities
    field_kernel<<<8192, 128, 0, stream>>>(points, w0p, bws0, w1p, bws1,
                                           w2p, bws2, out);
}
